// DensityGuidedCompressor_25108378812788
// MI455X (gfx1250) — matrix operand order repaired and checked by reading, not compiled, not run on hardware
//
#include <hip/hip_runtime.h>
#include <hip/hip_bf16.h>
#include <math.h>

// ---------------- problem constants ----------------
#define BATCH 8
#define NTOK  4096
#define DFEAT 1024
#define HDIM  1024
#define NQ    64
#define TOPK  1024
#define H2    2048   // 2*H for density MLP

typedef __attribute__((ext_vector_type(16))) __bf16          v16bf;
typedef __attribute__((ext_vector_type(16))) unsigned short  v16u;
typedef __attribute__((ext_vector_type(8)))  float           v8f;
typedef __attribute__((ext_vector_type(2)))  float           v2f;

// f32 -> bf16 round-to-nearest-even (bit trick), and back
__device__ __forceinline__ unsigned short f32_bf16_rne(float f) {
    unsigned u = __float_as_uint(f);
    unsigned r = u + 0x7FFFu + ((u >> 16) & 1u);
    return (unsigned short)(r >> 16);
}
__device__ __forceinline__ float bf16_f32(unsigned short h) {
    return __uint_as_float(((unsigned)h) << 16);
}

// ---------------------------------------------------------------------------
// Kernel 1: density bias per token: bias = relu(d*w1+b1) . w2 + b2
// one wave per token; weights staged in LDS
// ---------------------------------------------------------------------------
__global__ void k_density(const float* __restrict__ dens,
                          const float* __restrict__ w1,
                          const float* __restrict__ b1,
                          const float* __restrict__ w2,
                          const float* __restrict__ b2,
                          float* __restrict__ bias) {
    __shared__ float w1s[H2], b1s[H2], w2s[H2];
    int tid = threadIdx.x;                       // 256
    for (int i = tid; i < H2; i += 256) { w1s[i] = w1[i]; b1s[i] = b1[i]; w2s[i] = w2[i]; }
    __syncthreads();
    int lane = tid & 31, wid = tid >> 5;
    int token = blockIdx.x * 8 + wid;            // 4096 blocks * 8 waves = 32768 tokens
    float d = dens[token];
    float sum = 0.f;
    #pragma unroll 8
    for (int i = 0; i < 64; ++i) {
        int j = i * 32 + lane;
        float h = fmaf(d, w1s[j], b1s[j]);
        h = fmaxf(h, 0.f);
        sum = fmaf(h, w2s[j], sum);
    }
    for (int off = 16; off >= 1; off >>= 1) sum += __shfl_xor(sum, off, 32);
    if (lane == 0) bias[token] = sum + b2[0];
}

// ---------------------------------------------------------------------------
// Kernel 2: Wq[d,q] = sum_h key_w[d,h]*Q[q,h], scaled by 1/sqrt(H)=1/32.
// f32 WMMA 16x16x4. One wave per 16x16 tile; 4 independent acc chains.
// ---------------------------------------------------------------------------
__global__ void k_wq(const float* __restrict__ kw, const float* __restrict__ qe,
                     float* __restrict__ wq) {
    int tid = threadIdx.x;
    int lane = tid & 31, wid = tid >> 5;
    int w  = blockIdx.x * 8 + wid;               // 0..255 waves
    int dt = w >> 2, qt = w & 3;                 // 64 d-tiles x 4 q-tiles
    int lr = lane & 15, lk = (lane >> 4) * 2;
    const float* ap = kw + (size_t)(dt * 16 + lr) * HDIM + lk;   // A: key_w rows
    const float* bp = qe + (size_t)(qt * 16 + lr) * HDIM + lk;   // B: Q^T cols
    v8f acc0 = {}, acc1 = {}, acc2 = {}, acc3 = {};
    for (int k0 = 0; k0 < HDIM; k0 += 16) {
        v2f a0 = { ap[k0+ 0], ap[k0+ 1] }, b0 = { bp[k0+ 0], bp[k0+ 1] };
        v2f a1 = { ap[k0+ 4], ap[k0+ 5] }, b1 = { bp[k0+ 4], bp[k0+ 5] };
        v2f a2 = { ap[k0+ 8], ap[k0+ 9] }, b2 = { bp[k0+ 8], bp[k0+ 9] };
        v2f a3 = { ap[k0+12], ap[k0+13] }, b3 = { bp[k0+12], bp[k0+13] };
        acc0 = __builtin_amdgcn_wmma_f32_16x16x4_f32(false, a0, false, b0, (short)0, acc0, false, false);
        acc1 = __builtin_amdgcn_wmma_f32_16x16x4_f32(false, a1, false, b1, (short)0, acc1, false, false);
        acc2 = __builtin_amdgcn_wmma_f32_16x16x4_f32(false, a2, false, b2, (short)0, acc2, false, false);
        acc3 = __builtin_amdgcn_wmma_f32_16x16x4_f32(false, a3, false, b3, (short)0, acc3, false, false);
    }
    const float scale = 1.0f / 32.0f;            // 1/sqrt(H), key_b dropped (softmax-invariant)
    int rowbase = dt * 16 + 8 * (lane >> 4);
    #pragma unroll
    for (int v = 0; v < 8; ++v) {
        float val = (acc0[v] + acc1[v]) + (acc2[v] + acc3[v]);
        wq[(size_t)(rowbase + v) * NQ + qt * 16 + lr] = val * scale;
    }
}

// ---------------------------------------------------------------------------
// Kernel 3: swizzle Wq (f32) into bf16 hi/lo B-fragments for 16x16x32 WMMA.
// B layout (16-bit, 32x16): lane holds col = lane&15, K = (h<8 ? h : h+8) + 8*(lane>>4).
// ---------------------------------------------------------------------------
__global__ void k_wqswz(const float* __restrict__ wq,
                        unsigned short* __restrict__ hi,
                        unsigned short* __restrict__ lo) {
    int t = blockIdx.x * 256 + threadIdx.x;      // 0..4095 = 32 ksteps * 4 qt * 32 lanes
    int kstep = t >> 7;
    int qt    = (t >> 5) & 3;
    int lane  = t & 31;
    int col = qt * 16 + (lane & 15);
    int kb  = 8 * (lane >> 4);
    size_t base = ((size_t)(kstep * 4 + qt) * 32 + lane) * 16;
    #pragma unroll
    for (int h = 0; h < 16; ++h) {
        int K = kstep * 32 + kb + (h < 8 ? h : h + 8);
        float f = wq[(size_t)K * NQ + col];
        unsigned short hb = f32_bf16_rne(f);
        hi[base + h] = hb;
        lo[base + h] = f32_bf16_rne(f - bf16_f32(hb));
    }
}

// ---------------------------------------------------------------------------
// Kernel 4: main pass. s[b,n,q] = (tf[b,n,:] . Wq[:,q]) + bias[b,n]
// bf16 split-precision: Ah*Bh + Ah*Bl + Al*Bh, f32 accumulate.
// Wave = 16 token rows x all 64 queries (4 acc tiles). Block = 8 waves = 128 rows.
// A layout (16-bit, 16x32): lane holds row = lane&15,
//   K = (h<8 ? h : h+8) + 8*(lane>>4)  -> two contiguous 8-float runs at +0 and +16.
// ---------------------------------------------------------------------------
__global__ void k_scores(const float* __restrict__ tf,
                         const unsigned short* __restrict__ wqhi,
                         const unsigned short* __restrict__ wqlo,
                         const float* __restrict__ bias,
                         float* __restrict__ s) {
    int b  = blockIdx.y;
    int nt = blockIdx.x;
    int tid = threadIdx.x;
    int lane = tid & 31, wid = tid >> 5;
    int n0 = nt * 128 + wid * 16;
    int lr = lane & 15;
    int lh = lane >> 4;
    const float* arow = tf + ((size_t)b * NTOK + n0 + lr) * DFEAT + 8 * lh;
    v8f acc0 = {}, acc1 = {}, acc2 = {}, acc3 = {};
    for (int kk = 0; kk < DFEAT / 32; ++kk) {
        const float* ap = arow + kk * 32;
        float4 x0 = *(const float4*)(ap);
        float4 x1 = *(const float4*)(ap + 4);
        float4 x2 = *(const float4*)(ap + 16);
        float4 x3 = *(const float4*)(ap + 20);
        float af[16] = { x0.x,x0.y,x0.z,x0.w, x1.x,x1.y,x1.z,x1.w,
                         x2.x,x2.y,x2.z,x2.w, x3.x,x3.y,x3.z,x3.w };
        v16u hiu, lou;
        #pragma unroll
        for (int h = 0; h < 16; ++h) {
            unsigned short hb = f32_bf16_rne(af[h]);
            hiu[h] = hb;
            lou[h] = f32_bf16_rne(af[h] - bf16_f32(hb));
        }
        v16bf Ah = __builtin_bit_cast(v16bf, hiu);
        v16bf Al = __builtin_bit_cast(v16bf, lou);
        #define QT_STEP(i, ACC)                                                              \
        {                                                                                    \
            v16bf Bh = *(const v16bf*)(wqhi + ((size_t)(kk * 4 + (i)) * 32 + lane) * 16);    \
            v16bf Bl = *(const v16bf*)(wqlo + ((size_t)(kk * 4 + (i)) * 32 + lane) * 16);    \
            ACC = __builtin_amdgcn_wmma_f32_16x16x32_bf16(false, Ah, false, Bh, (short)0, ACC, false, false); \
            ACC = __builtin_amdgcn_wmma_f32_16x16x32_bf16(false, Ah, false, Bl, (short)0, ACC, false, false); \
            ACC = __builtin_amdgcn_wmma_f32_16x16x32_bf16(false, Al, false, Bh, (short)0, ACC, false, false); \
        }
        QT_STEP(0, acc0) QT_STEP(1, acc1) QT_STEP(2, acc2) QT_STEP(3, acc3)
        #undef QT_STEP
    }
    // C/D layout: VGPR v, lane: row = v + 8*(lane>>4), col = lane&15
    #pragma unroll
    for (int v = 0; v < 8; ++v) {
        int n = n0 + v + 8 * lh;
        float bv = bias[b * NTOK + n];
        float* srow = s + ((size_t)b * NTOK + n) * NQ;
        srow[ 0 + lr] = acc0[v] + bv;
        srow[16 + lr] = acc1[v] + bv;
        srow[32 + lr] = acc2[v] + bv;
        srow[48 + lr] = acc3[v] + bv;
    }
}

// ---------------------------------------------------------------------------
// Kernel 5: per-(b,q) column stats: C = max_n s + log(sum_n exp(s - max))
// ---------------------------------------------------------------------------
__global__ void k_colstats(const float* __restrict__ s, float* __restrict__ C) {
    int q = blockIdx.x;                          // 64
    int b = blockIdx.y;                          // 8
    int tid = threadIdx.x;                       // 256
    float vals[16];
    float m = -3.4e38f;
    #pragma unroll
    for (int i = 0; i < 16; ++i) {
        int n = tid + i * 256;
        float v = s[((size_t)b * NTOK + n) * NQ + q];
        vals[i] = v;
        m = fmaxf(m, v);
    }
    __shared__ float red[256];
    red[tid] = m; __syncthreads();
    for (int off = 128; off >= 1; off >>= 1) {
        if (tid < off) red[tid] = fmaxf(red[tid], red[tid + off]);
        __syncthreads();
    }
    float M = red[0];
    __syncthreads();
    float zs = 0.f;
    #pragma unroll
    for (int i = 0; i < 16; ++i) zs += expf(vals[i] - M);
    red[tid] = zs; __syncthreads();
    for (int off = 128; off >= 1; off >>= 1) {
        if (tid < off) red[tid] += red[tid + off];
        __syncthreads();
    }
    if (tid == 0) C[b * NQ + q] = M + logf(red[0]);
}

// ---------------------------------------------------------------------------
// Kernel 6: r[b,n] = max_q (s[b,n,q] - C[b,q])   (== log importance; monotonic)
// ---------------------------------------------------------------------------
__global__ void k_rowmax(const float* __restrict__ s, const float* __restrict__ C,
                         float* __restrict__ r) {
    int g = blockIdx.x * 256 + threadIdx.x;      // 0..32767
    int b = g >> 12;
    const float* srow = s + (size_t)g * NQ;
    const float* Cb = C + b * NQ;
    float mx = -3.4e38f;
    #pragma unroll 8
    for (int q = 0; q < NQ; ++q) mx = fmaxf(mx, srow[q] - Cb[q]);
    r[g] = mx;
}

// ---------------------------------------------------------------------------
// Kernel 7: exact stable top-1024 of 4096 per batch, indices ascending.
// 12-bit histogram + suffix scan finds threshold bin; exact tie-rank inside it.
// ---------------------------------------------------------------------------
__global__ void k_select(const float* __restrict__ r, int* __restrict__ idx) {
    const int b = blockIdx.x;
    const int tid = threadIdx.x;                 // 0..1023
    __shared__ unsigned keys[NTOK];
    __shared__ unsigned hist[NTOK];
    __shared__ unsigned gsum[1024];
    __shared__ unsigned sel[NTOK];
    __shared__ unsigned s_bstar, s_need;

    for (int i = tid; i < NTOK; i += 1024) hist[i] = 0u;
    __syncthreads();
    for (int i = tid; i < NTOK; i += 1024) {
        unsigned u = __float_as_uint(r[b * NTOK + i]);
        unsigned k = (u & 0x80000000u) ? ~u : (u | 0x80000000u);  // order-preserving
        keys[i] = k;
        atomicAdd(&hist[k >> 20], 1u);
    }
    __syncthreads();

    // suffix sums over 4096 bins: thread owns bins 4t..4t+3
    unsigned h0 = hist[4*tid+0], h1 = hist[4*tid+1], h2 = hist[4*tid+2], h3 = hist[4*tid+3];
    unsigned ls3 = h3, ls2 = h3 + h2, ls1 = ls2 + h1, ls0 = ls1 + h0;
    gsum[tid] = ls0;
    __syncthreads();
    for (int off = 1; off < 1024; off <<= 1) {   // inclusive suffix scan
        unsigned x = (tid + off < 1024) ? gsum[tid + off] : 0u;
        __syncthreads();
        gsum[tid] += x;
        __syncthreads();
    }
    {
        unsigned tail = gsum[tid] - ls0;         // sum over groups > t
        unsigned suf[4]  = { tail + ls0, tail + ls1, tail + ls2, tail + ls3 };
        unsigned sufn[4] = { suf[1], suf[2], suf[3], tail };
        #pragma unroll
        for (int j = 0; j < 4; ++j) {
            if (suf[j] >= TOPK && sufn[j] < TOPK) {  // unique bin
                s_bstar = (unsigned)(4 * tid + j);
                s_need  = TOPK - sufn[j];
            }
        }
    }
    __syncthreads();
    const unsigned bstar = s_bstar, need = s_need;

    for (int i = tid; i < NTOK; i += 1024) {
        unsigned k = keys[i];
        unsigned bin = k >> 20;
        unsigned sl = 0u;
        if (bin > bstar) sl = 1u;
        else if (bin == bstar) {                 // stable (value desc, index asc) tie rank
            unsigned cnt = 0u;
            for (int m = 0; m < NTOK; ++m) {
                unsigned km = keys[m];
                if ((km >> 20) == bstar && (km > k || (km == k && m < i))) cnt++;
            }
            sl = (cnt < need) ? 1u : 0u;
        }
        sel[i] = sl;
    }
    __syncthreads();

    // exclusive prefix over sel -> output slots (ascending index order)
    unsigned e0 = sel[4*tid+0], e1 = sel[4*tid+1], e2 = sel[4*tid+2], e3 = sel[4*tid+3];
    unsigned p1 = e0, p2 = e0 + e1, p3 = e0 + e1 + e2, tot = p3 + e3;
    gsum[tid] = tot;
    __syncthreads();
    for (int off = 1; off < 1024; off <<= 1) {   // inclusive prefix scan
        unsigned x = (tid >= off) ? gsum[tid - off] : 0u;
        __syncthreads();
        gsum[tid] += x;
        __syncthreads();
    }
    unsigned base = gsum[tid] - tot;
    unsigned pos[4] = { base, base + p1, base + p2, base + p3 };
    #pragma unroll
    for (int j = 0; j < 4; ++j) {
        int i = 4 * tid + j;
        if (sel[i]) idx[b * TOPK + pos[j]] = i;
    }
}

// ---------------------------------------------------------------------------
// Kernel 8: gather selected rows. One block per (b,k); float4 copies.
// ---------------------------------------------------------------------------
__global__ void k_gather(const float* __restrict__ tf, const int* __restrict__ idx,
                         float* __restrict__ out) {
    int b = blockIdx.y;
    int k = blockIdx.x;
    int row = idx[b * TOPK + k];
    const float4* src = (const float4*)(tf + ((size_t)b * NTOK + row) * DFEAT);
    float4* dst = (float4*)(out + ((size_t)b * TOPK + k) * DFEAT);
    dst[threadIdx.x] = src[threadIdx.x];         // 256 threads * 4 floats = 1024
}

// ---------------------------------------------------------------------------
extern "C" void kernel_launch(void* const* d_in, const int* in_sizes, int n_in,
                              void* d_out, int out_size, void* d_ws, size_t ws_size,
                              hipStream_t stream) {
    const float* tf   = (const float*)d_in[0];   // [8,4096,1024]
    const float* dens = (const float*)d_in[1];   // [8,4096]
    const float* qe   = (const float*)d_in[2];   // [64,1024]
    const float* kw   = (const float*)d_in[3];   // [1024,1024]
    // d_in[4] = key_b: per-query constant -> cancels in softmax over tokens; unused.
    const float* w1   = (const float*)d_in[5];   // [1,2048]
    const float* b1   = (const float*)d_in[6];   // [2048]
    const float* w2   = (const float*)d_in[7];   // [2048,1]
    const float* b2   = (const float*)d_in[8];   // [1]
    float* out = (float*)d_out;                  // [8,1024,1024]

    // workspace layout (~8.8 MiB)
    float*          wq   = (float*)d_ws;                          // 65536 f32
    unsigned short* wqhi = (unsigned short*)(wq + 65536);         // 65536 bf16
    unsigned short* wqlo = wqhi + 65536;                          // 65536 bf16
    float*          bias = (float*)(wqlo + 65536);                // 32768 f32
    float*          s    = bias + 32768;                          // 2097152 f32
    float*          Cst  = s + (size_t)BATCH * NTOK * NQ;         // 512 f32
    float*          rr   = Cst + 512;                             // 32768 f32
    int*            idx  = (int*)(rr + 32768);                    // 8192 i32

    k_density <<<dim3(4096),    dim3(256),  0, stream>>>(dens, w1, b1, w2, b2, bias);
    k_wq      <<<dim3(32),      dim3(256),  0, stream>>>(kw, qe, wq);
    k_wqswz   <<<dim3(16),      dim3(256),  0, stream>>>(wq, wqhi, wqlo);
    k_scores  <<<dim3(32, 8),   dim3(256),  0, stream>>>(tf, wqhi, wqlo, bias, s);
    k_colstats<<<dim3(64, 8),   dim3(256),  0, stream>>>(s, Cst);
    k_rowmax  <<<dim3(128),     dim3(256),  0, stream>>>(s, Cst, rr);
    k_select  <<<dim3(8),       dim3(1024), 0, stream>>>(rr, idx);
    k_gather  <<<dim3(1024, 8), dim3(256),  0, stream>>>(tf, idx, out);
}
